// CausalTransformer_81776177316304
// MI455X (gfx1250) — matrix operand order repaired and checked by reading, not compiled, not run on hardware
//
#include <hip/hip_runtime.h>
#include <hip/hip_bf16.h>
#include <math.h>

typedef __attribute__((ext_vector_type(16))) _Float16 v16h;
typedef __attribute__((ext_vector_type(8)))  _Float16 v8h;
typedef __attribute__((ext_vector_type(8)))  float    v8f;

#define B_  2
#define S_  1024
#define E_  768
#define H_  12
#define L_  4
#define FF_ 2048
#define D_  64          // head dim
#define M_  (B_*S_)     // 2048 token rows

// ---------------------------------------------------------------------------
// Optional CDNA5 Tensor Data Mover staging (guarded; 5-arg on clang-22/ROCm7.2,
// 6-arg on clang-23/amdgpu-toolchain). Builds a 2-D D# per ISA 08_async_tensor.
// ---------------------------------------------------------------------------
#if __has_builtin(__builtin_amdgcn_tensor_load_to_lds)
#define HAVE_TDM 1
typedef __attribute__((ext_vector_type(4))) unsigned int u32x4;
typedef __attribute__((ext_vector_type(4))) int          i32x4;
typedef __attribute__((ext_vector_type(8))) int          i32x8;

__device__ __forceinline__ void tdm_load_2d_f16(unsigned lds_off, const void* gaddr,
                                                unsigned tile_w, unsigned tile_h,
                                                unsigned tensor_w, unsigned tensor_h,
                                                unsigned stride_elems) {
  unsigned long long ga = (unsigned long long)gaddr;
  u32x4 g0;
  g0[0] = 1u;                                            // count=1, user mode
  g0[1] = lds_off;                                       // lds_addr [63:32]
  g0[2] = (unsigned)(ga & 0xffffffffu);                  // global_addr lo
  g0[3] = (unsigned)((ga >> 32) & 0x01ffffffu) | (2u << 30); // addr hi + type=2
  i32x8 g1;
  g1[0] = (int)(1u << 16);                               // data_size=1 (2 bytes)
  g1[1] = (int)((tensor_w & 0xffffu) << 16);             // tensor_dim0[15:0]
  g1[2] = (int)((tensor_w >> 16) & 0xffffu) |
          (int)((tensor_h & 0xffffu) << 16);             // dim0 hi | tensor_dim1 lo
  g1[3] = (int)((tensor_h >> 16) & 0xffffu) |
          (int)((tile_w & 0xffffu) << 16);               // dim1 hi | tile_dim0
  g1[4] = (int)(tile_h & 0xffffu);                       // tile_dim1 (tile_dim2=0)
  g1[5] = (int)stride_elems;                             // tensor_dim0_stride lo
  g1[6] = 0;                                             // stride0 hi | stride1 lo
  g1[7] = 0;
  i32x4 z4 = {0, 0, 0, 0};
#if __clang_major__ >= 23
  i32x8 z8 = {0, 0, 0, 0, 0, 0, 0, 0};
  __builtin_amdgcn_tensor_load_to_lds(g0, g1, z4, z4, z8, 0);
#else
  __builtin_amdgcn_tensor_load_to_lds(g0, g1, z4, z4, 0);
#endif
}
#endif

// ---------------------------------------------------------------------------
// WMMA fragment helpers (V_WMMA_F32_16X16X32_F16 lane layouts per CDNA5 ISA)
// ---------------------------------------------------------------------------
__device__ __forceinline__ v16h load_a_frag16(const _Float16* __restrict__ A, int lda,
                                              int row0, int k0, int lane) {
  const _Float16* p = A + (size_t)(row0 + (lane & 15)) * lda + k0 + ((lane & 16) ? 8 : 0);
  v8h lo = *(const v8h*)p;
  v8h hi = *(const v8h*)(p + 16);
  return __builtin_shufflevector(lo, hi, 0,1,2,3,4,5,6,7,8,9,10,11,12,13,14,15);
}

__device__ __forceinline__ v16h load_b_frag16(const _Float16* __restrict__ W, int ldw,
                                              int n0, int k0, int lane) {
  const _Float16* p = W + (size_t)(n0 + (lane & 15)) * ldw + k0 + ((lane & 16) ? 8 : 0);
  v8h lo = *(const v8h*)p;
  v8h hi = *(const v8h*)(p + 16);
  return __builtin_shufflevector(lo, hi, 0,1,2,3,4,5,6,7,8,9,10,11,12,13,14,15);
}

// A fragment from f32 source (LDS-resident softmax probabilities).
__device__ __forceinline__ v16h load_a_frag_f32(const float* __restrict__ A, int lda,
                                                int row0, int k0, int lane) {
  const float* p = A + (size_t)(row0 + (lane & 15)) * lda + k0 + ((lane & 16) ? 8 : 0);
  v16h f;
#pragma unroll
  for (int i = 0; i < 8; ++i) f[i] = (_Float16)p[i];
#pragma unroll
  for (int i = 0; i < 8; ++i) f[8 + i] = (_Float16)p[16 + i];
  return f;
}

__device__ __forceinline__ v8f wmma_f16(v16h a, v16h b, v8f c) {
  return __builtin_amdgcn_wmma_f32_16x16x32_f16(false, a, false, b, (short)0, c,
                                                false, false);
}

// ---------------------------------------------------------------------------
// Bulk f32 -> f16 conversion (weights, once per launch)
// ---------------------------------------------------------------------------
__global__ __launch_bounds__(256)
void cvt_kernel(const float* __restrict__ src, _Float16* __restrict__ dst, int n) {
  for (int i = (blockIdx.x * 256 + threadIdx.x) * 4; i < n; i += gridDim.x * 256 * 4) {
    float4 v = *(const float4*)(src + i);
    dst[i + 0] = (_Float16)v.x;
    dst[i + 1] = (_Float16)v.y;
    dst[i + 2] = (_Float16)v.z;
    dst[i + 3] = (_Float16)v.w;
  }
}

// ---------------------------------------------------------------------------
// V transpose: vt[b][h][n][k] = qkv[b][k][2E + h*64 + n]  (f16, ld = S)
// Turns P@V B-fragments into contiguous b128 loads.
// ---------------------------------------------------------------------------
__global__ __launch_bounds__(256)
void vtrans_kernel(const _Float16* __restrict__ qkv, _Float16* __restrict__ vt) {
  int bh = blockIdx.y;                  // b*H + h
  int b  = bh / H_, hh = bh % H_;
  int n  = blockIdx.x;                  // 0..63
  const _Float16* src = qkv + (size_t)b * S_ * (3 * E_) + 2 * E_ + hh * D_ + n;
  _Float16* dst = vt + ((size_t)bh * D_ + n) * S_;
  for (int k = threadIdx.x; k < S_; k += 256)
    dst[k] = src[(size_t)k * (3 * E_)];
}

// ---------------------------------------------------------------------------
// Dual positional encoding + zero padding; writes f32 residual and f16 operand.
// ---------------------------------------------------------------------------
__global__ __launch_bounds__(256)
void pe_kernel(const float* __restrict__ x, const float* __restrict__ tpe,
               const int* __restrict__ ttp, const int* __restrict__ rtp,
               float* __restrict__ h32, _Float16* __restrict__ h16) {
  int s = blockIdx.x % S_;
  int b = blockIdx.x / S_;
  int nt = *ttp + 1;
  int rt = *rtp;
  int valid = rt * nt;
  float*    hp = h32 + (size_t)blockIdx.x * E_;
  _Float16* hh = h16 + (size_t)blockIdx.x * E_;
  if (s >= valid) {
    for (int e = threadIdx.x; e < E_; e += blockDim.x) { hp[e] = 0.0f; hh[e] = (_Float16)0.0f; }
    return;
  }
  int t = s / nt, j = s % nt;
  const float* xp = x + ((size_t)b * S_ + s) * E_;
  const float* tp = tpe + (size_t)j * E_;
  const float negln = -logf(10000.0f) / (float)E_;
  for (int e = threadIdx.x; e < E_; e += blockDim.x) {
    int pair = e >> 1;
    float div = expf((float)(2 * pair) * negln);
    float ang = (float)t * div;
    float pe = (e & 1) ? cosf(ang) : sinf(ang);
    float v = xp[e] + pe + tp[e];
    hp[e] = v;
    hh[e] = (_Float16)v;
  }
}

// ---------------------------------------------------------------------------
// C[M x N] = act(A[M x K] @ W[N x K]^T + bias). f16 operands, f32 accumulate.
// 8 waves; block tile 32(M) x 256(N); wave tile 16 x 64. Software-pipelined K.
// Optional TDM staging of the W panel into LDS (async-tensor demo / L2 warm).
// ---------------------------------------------------------------------------
__global__ __launch_bounds__(256)
void gemm_wmma_kernel(const _Float16* __restrict__ A, const _Float16* __restrict__ W,
                      const float* __restrict__ bias,
                      float* __restrict__ Cf, _Float16* __restrict__ Ch,
                      int N, int K, int act) {
  int lane  = threadIdx.x & 31;
  int wave  = threadIdx.x >> 5;
  int waveM = wave >> 2;                 // 0..1
  int waveN = wave & 3;                  // 0..3
  int row0  = blockIdx.y * 32 + waveM * 16;
  int nBase = blockIdx.x * 256 + waveN * 64;

#ifdef HAVE_TDM
  __shared__ _Float16 stage[64 * 64];    // 8 KB TDM staging tile
  if (threadIdx.x < 32) {                // one wave issues the tile DMA
    tdm_load_2d_f16((unsigned)(unsigned long long)&stage[0],
                    W + (size_t)(blockIdx.x * 256) * K,
                    /*tile_w=*/64, /*tile_h=*/64,
                    /*tensor_w=*/(unsigned)K, /*tensor_h=*/(unsigned)N,
                    /*stride=*/(unsigned)K);
  }
#endif

  v8f a0 = {}, a1 = {}, a2 = {}, a3 = {};

  // Software pipeline: preload next K-step while current WMMAs execute.
  v16h fa = load_a_frag16(A, K, row0, 0, lane);
  v16h b0 = load_b_frag16(W, K, nBase + 0,  0, lane);
  v16h b1 = load_b_frag16(W, K, nBase + 16, 0, lane);
  v16h b2 = load_b_frag16(W, K, nBase + 32, 0, lane);
  v16h b3 = load_b_frag16(W, K, nBase + 48, 0, lane);

  for (int k = 32; k < K; k += 32) {
    v16h na = load_a_frag16(A, K, row0, k, lane);
    v16h c0 = load_b_frag16(W, K, nBase + 0,  k, lane);
    v16h c1 = load_b_frag16(W, K, nBase + 16, k, lane);
    v16h c2 = load_b_frag16(W, K, nBase + 32, k, lane);
    v16h c3 = load_b_frag16(W, K, nBase + 48, k, lane);
    if (k + 32 < K)   // warm next A slice (global_prefetch_b8)
      __builtin_prefetch(A + (size_t)(row0 + (lane & 15)) * K + k + 32, 0, 3);
    a0 = wmma_f16(fa, b0, a0);
    a1 = wmma_f16(fa, b1, a1);
    a2 = wmma_f16(fa, b2, a2);
    a3 = wmma_f16(fa, b3, a3);
    fa = na; b0 = c0; b1 = c1; b2 = c2; b3 = c3;
  }
  a0 = wmma_f16(fa, b0, a0);
  a1 = wmma_f16(fa, b1, a1);
  a2 = wmma_f16(fa, b2, a2);
  a3 = wmma_f16(fa, b3, a3);

  int rbase  = row0 + ((lane & 16) ? 8 : 0);
  int cshift = lane & 15;
  v8f accs[4] = {a0, a1, a2, a3};
#pragma unroll
  for (int j = 0; j < 4; ++j) {
    int col = nBase + j * 16 + cshift;
    float bv = bias[col];
#pragma unroll
    for (int i = 0; i < 8; ++i) {
      float v = accs[j][i] + bv;
      if (act) v = 0.5f * v * (1.0f + erff(v * 0.70710678118f));   // exact GELU
      if (Cf) Cf[(size_t)(rbase + i) * N + col] = v;
      if (Ch) Ch[(size_t)(rbase + i) * N + col] = (_Float16)v;
    }
  }

#ifdef HAVE_TDM
#if __has_builtin(__builtin_amdgcn_s_wait_tensorcnt)
  if (threadIdx.x < 32) __builtin_amdgcn_s_wait_tensorcnt(0);
#endif
#endif
}

// ---------------------------------------------------------------------------
// Attention: one block per (b, head, 16-query tile). 256 threads = 8 waves.
// ---------------------------------------------------------------------------
__global__ __launch_bounds__(256)
void attn_kernel(const _Float16* __restrict__ qkv, const _Float16* __restrict__ vt,
                 float* __restrict__ out,
                 const int* __restrict__ ttp, const int* __restrict__ rtp) {
  __shared__ float sc[16][S_];           // 16 x 1024 f32 = 64 KB

  int lane  = threadIdx.x & 31;
  int wave  = threadIdx.x >> 5;
  int qBase = blockIdx.x * 16;
  int hh    = blockIdx.y;
  int b     = blockIdx.z;

  const _Float16* base = qkv + (size_t)b * S_ * (3 * E_);
  const _Float16* Q  = base + hh * D_;            // ld = 3E
  const _Float16* Km = base + E_ + hh * D_;       // ld = 3E
  const _Float16* VtB = vt + (size_t)(b * H_ + hh) * D_ * S_;  // ld = S

  v16h q0 = load_a_frag16(Q, 3 * E_, qBase, 0, lane);
  v16h q1 = load_a_frag16(Q, 3 * E_, qBase, 32, lane);

  const float scale = 0.125f;            // 1/sqrt(64)

  // ---- Phase 1: S = scale * Q @ K^T -> LDS
  for (int kt = wave; kt < S_ / 16; kt += 8) {
    v8f s = {};
    s = wmma_f16(q0, load_b_frag16(Km, 3 * E_, kt * 16, 0, lane), s);
    s = wmma_f16(q1, load_b_frag16(Km, 3 * E_, kt * 16, 32, lane), s);
    int r0 = (lane & 16) ? 8 : 0;
    int c  = kt * 16 + (lane & 15);
#pragma unroll
    for (int i = 0; i < 8; ++i) sc[r0 + i][c] = s[i] * scale;
  }
  __syncthreads();

  // ---- Phase 2: block-causal mask + softmax (wave handles 2 rows)
  int tt = *ttp, ntk = *rtp;
  int bs = tt + 1;
  int lim = ntk * bs;
  const float NEG_INF = -__builtin_inff();

  for (int rr = 0; rr < 2; ++rr) {
    int r  = wave * 2 + rr;
    int ig = qBase + r;
    int ib = ig / bs, im = ig % bs;
    float m = NEG_INF;
    for (int c = lane; c < S_; c += 32) {
      int jb = c / bs, jm = c % bs;
      bool same = (ib == jb) && (ib < ntk) && (im >= jm);
      bool sv   = (jm == 0) && (jb < ntk) && (ig > c) && (ig < lim);
      float v = (same || sv) ? sc[r][c] : NEG_INF;
      sc[r][c] = v;
      m = fmaxf(m, v);
    }
#pragma unroll
    for (int o = 16; o > 0; o >>= 1) m = fmaxf(m, __shfl_xor(m, o, 32));
    bool dead = (m == NEG_INF);          // fully-masked row -> zeros (nan_to_num)
    float sum = 0.0f;
    for (int c = lane; c < S_; c += 32) {
      float e = dead ? 0.0f : expf(sc[r][c] - m);
      sc[r][c] = e;
      sum += e;
    }
#pragma unroll
    for (int o = 16; o > 0; o >>= 1) sum += __shfl_xor(sum, o, 32);
    float inv = (sum > 0.0f) ? 1.0f / sum : 0.0f;
    for (int c = lane; c < S_; c += 32) sc[r][c] *= inv;
  }
  __syncthreads();

  // ---- Phase 3: O = P @ V, split-K across 8 waves + LDS partial reduction.
  {
    int n0   = (wave & 3) * 16;          // 16-wide slice of d=64
    int kBeg = (wave >> 2) * (S_ / 2);   // key-range half
    v8f o = {};
    for (int k = kBeg; k < kBeg + S_ / 2; k += 32) {
      v16h a  = load_a_frag_f32(&sc[0][0], S_, 0, k, lane);
      v16h bb = load_b_frag16(VtB, S_, n0, k, lane);
      o = wmma_f16(a, bb, o);
    }
    __syncthreads();                     // all probs consumed; sc reusable
    if (wave >= 4) {
      float* dst = &sc[wave - 4][0];     // deposit partial 16x16 tile
#pragma unroll
      for (int i = 0; i < 8; ++i) dst[i * 32 + lane] = o[i];
    }
    __syncthreads();
    if (wave < 4) {
      const float* src = &sc[wave][0];
#pragma unroll
      for (int i = 0; i < 8; ++i) o[i] += src[i * 32 + lane];
      int r0 = (lane & 16) ? 8 : 0;
      int c  = n0 + (lane & 15);
#pragma unroll
      for (int i = 0; i < 8; ++i)
        out[((size_t)b * S_ + qBase + r0 + i) * E_ + hh * D_ + c] = o[i];
    }
  }
}

// ---------------------------------------------------------------------------
// out = LayerNorm(a + r) * w + bias ; optional NaN/Inf sanitize (nan_to_num).
// ---------------------------------------------------------------------------
__global__ __launch_bounds__(256)
void add_ln_kernel(const float* __restrict__ a, const float* __restrict__ r,
                   const float* __restrict__ w, const float* __restrict__ bias,
                   float* __restrict__ out32, _Float16* __restrict__ out16,
                   int sanitize) {
  __shared__ float red[8];
  int row  = blockIdx.x;
  int lane = threadIdx.x & 31;
  int wave = threadIdx.x >> 5;
  const float* ap = a + (size_t)row * E_;
  const float* rp = r ? r + (size_t)row * E_ : nullptr;

  float v[3];
  float sum = 0.0f;
#pragma unroll
  for (int i = 0; i < 3; ++i) {
    int e = threadIdx.x + i * 256;
    float t = ap[e] + (rp ? rp[e] : 0.0f);
    v[i] = t;
    sum += t;
  }
#pragma unroll
  for (int o = 16; o > 0; o >>= 1) sum += __shfl_xor(sum, o, 32);
  if (lane == 0) red[wave] = sum;
  __syncthreads();
  float tot = 0.0f;
#pragma unroll
  for (int i = 0; i < 8; ++i) tot += red[i];
  float mu = tot / (float)E_;
  __syncthreads();

  float s2 = 0.0f;
#pragma unroll
  for (int i = 0; i < 3; ++i) { float d = v[i] - mu; s2 += d * d; }
#pragma unroll
  for (int o = 16; o > 0; o >>= 1) s2 += __shfl_xor(s2, o, 32);
  if (lane == 0) red[wave] = s2;
  __syncthreads();
  float tv = 0.0f;
#pragma unroll
  for (int i = 0; i < 8; ++i) tv += red[i];
  float inv = rsqrtf(tv / (float)E_ + 1e-5f);

#pragma unroll
  for (int i = 0; i < 3; ++i) {
    int e = threadIdx.x + i * 256;
    float o2 = (v[i] - mu) * inv * w[e] + bias[e];
    if (sanitize) {
      if (o2 != o2) o2 = 0.0f;
      else if (fabsf(o2) == __builtin_inff())
        o2 = (o2 > 0.0f) ? 3.4028235e38f : -3.4028235e38f;
    }
    out32[(size_t)row * E_ + e] = o2;
    if (out16) out16[(size_t)row * E_ + e] = (_Float16)o2;
  }
}

// ---------------------------------------------------------------------------
extern "C" void kernel_launch(void* const* d_in, const int* in_sizes, int n_in,
                              void* d_out, int out_size, void* d_ws, size_t ws_size,
                              hipStream_t stream) {
  (void)in_sizes; (void)n_in; (void)out_size; (void)ws_size;
  const float* x    = (const float*)d_in[0];
  /* d_in[1] = padding_mask (unused, matching reference) */
  const float* tpe  = (const float*)d_in[2];
  const float* Wqkv = (const float*)d_in[3];
  const float* bqkv = (const float*)d_in[4];
  const float* W1   = (const float*)d_in[5];
  const float* b1   = (const float*)d_in[6];
  const float* W2   = (const float*)d_in[7];
  const float* b2   = (const float*)d_in[8];
  const float* ln1w = (const float*)d_in[9];
  const float* ln1b = (const float*)d_in[10];
  const float* ln2w = (const float*)d_in[11];
  const float* ln2b = (const float*)d_in[12];
  const float* lnfw = (const float*)d_in[13];
  const float* lnfb = (const float*)d_in[14];
  const int*   ttp  = (const int*)d_in[15];
  const int*   rtp  = (const int*)d_in[16];

  // ---- Workspace layout (all offsets 32B-aligned) ----
  char* p = (char*)d_ws;
  float*    h32   = (float*)p;      p += (size_t)M_ * E_ * 4;
  float*    tmp32 = (float*)p;      p += (size_t)M_ * E_ * 4;
  _Float16* h16   = (_Float16*)p;   p += (size_t)M_ * E_ * 2;
  _Float16* qkv16 = (_Float16*)p;   p += (size_t)M_ * 3 * E_ * 2;   // also ffn hidden
  _Float16* vt16  = (_Float16*)p;   p += (size_t)B_ * H_ * D_ * S_ * 2;
  _Float16* wq16  = (_Float16*)p;   p += (size_t)L_ * 3 * E_ * E_ * 2;
  _Float16* w116  = (_Float16*)p;   p += (size_t)L_ * FF_ * E_ * 2;
  _Float16* w216  = (_Float16*)p;   p += (size_t)L_ * E_ * FF_ * 2;

  // ---- Weight downconvert (once per launch) ----
  cvt_kernel<<<dim3(2048), dim3(256), 0, stream>>>(Wqkv, wq16, L_ * 3 * E_ * E_);
  cvt_kernel<<<dim3(2048), dim3(256), 0, stream>>>(W1,   w116, L_ * FF_ * E_);
  cvt_kernel<<<dim3(2048), dim3(256), 0, stream>>>(W2,   w216, L_ * E_ * FF_);

  pe_kernel<<<dim3(M_), dim3(256), 0, stream>>>(x, tpe, ttp, rtp, h32, h16);

  for (int l = 0; l < L_; ++l) {
    // QKV projection -> f16
    gemm_wmma_kernel<<<dim3(3 * E_ / 256, M_ / 32), dim3(256), 0, stream>>>(
        h16, wq16 + (size_t)l * 3 * E_ * E_, bqkv + (size_t)l * 3 * E_,
        nullptr, qkv16, 3 * E_, E_, 0);

    // V transpose for fast P@V fragments
    vtrans_kernel<<<dim3(D_, B_ * H_), dim3(256), 0, stream>>>(qkv16, vt16);

    // Attention -> tmp32
    attn_kernel<<<dim3(S_ / 16, H_, B_), dim3(256), 0, stream>>>(qkv16, vt16, tmp32, ttp, rtp);

    // h = LN1(h + attn)
    add_ln_kernel<<<dim3(M_), dim3(256), 0, stream>>>(
        h32, tmp32, ln1w + (size_t)l * E_, ln1b + (size_t)l * E_, h32, h16, 0);

    // FFN up + GELU -> f16 (reuse qkv16)
    gemm_wmma_kernel<<<dim3(FF_ / 256, M_ / 32), dim3(256), 0, stream>>>(
        h16, w116 + (size_t)l * FF_ * E_, b1 + (size_t)l * FF_,
        nullptr, qkv16, FF_, E_, 1);

    // FFN down -> f32 (residual)
    gemm_wmma_kernel<<<dim3(E_ / 256, M_ / 32), dim3(256), 0, stream>>>(
        qkv16, w216 + (size_t)l * E_ * FF_, b2 + (size_t)l * E_,
        tmp32, nullptr, E_, FF_, 0);

    // h = nan_to_num(LN2(h + ff))
    add_ln_kernel<<<dim3(M_), dim3(256), 0, stream>>>(
        h32, tmp32, ln2w + (size_t)l * E_, ln2b + (size_t)l * E_, h32, h16, 1);
  }

  // Final LayerNorm -> d_out
  add_ln_kernel<<<dim3(M_), dim3(256), 0, stream>>>(
      h32, nullptr, lnfw, lnfb, (float*)d_out, nullptr, 0);
}
